// LEAM_67645734912487
// MI455X (gfx1250) — matrix operand order repaired and checked by reading, not compiled, not run on hardware
//
#include <hip/hip_runtime.h>
#include <hip/hip_bf16.h>

// ---- problem constants (from reference setup_inputs) ----
#define BB 128
#define LL 1024
#define EE 300
#define CC 20
#define NG 55

typedef __attribute__((ext_vector_type(16))) __bf16 v16bf;
typedef __attribute__((ext_vector_type(8)))  __bf16 v8bf;
typedef __attribute__((ext_vector_type(8)))  float  v8f;

// ---- LDS layout (bytes) ----
// region0: max( Bmat hi+lo = 2*48*320*2 = 61440 , cwb = 55*20*32*2 = 70400 )
#define OFF_B0    0
#define R0_BYTES  70400
#define OFF_G     (OFF_B0 + R0_BYTES)            // Gpad: (27+1024+27) x 32 bf16 = 68992
#define GPAD_ROWS (LL + 54)
#define OFF_P     (OFF_G + GPAD_ROWS*32*2)       // P: 1024 x 20 f32 = 81920
#define OFF_MISC  (OFF_P + LL*CC*4)              // invn[32] + redD[32] + redN[32]
#define SMEM_BYTES (OFF_MISC + 3*32*4)

__device__ __forceinline__ float4 ld4_pred(const float* p, bool ok) {
    float4 r;
    if (ok) r = *(const float4*)p;
    else    r = make_float4(0.f, 0.f, 0.f, 0.f);
    return r;
}

__device__ __forceinline__ v16bf join16(v8bf a, v8bf b) {
    v16bf r;
#pragma unroll
    for (int i = 0; i < 8; ++i) { r[i] = a[i]; r[i + 8] = b[i]; }
    return r;
}

__global__ __launch_bounds__(512, 1)
void leam_fused_kernel(const float* __restrict__ x_emb,   // [B,L,E]
                       const float* __restrict__ x_mask,  // [B,L]
                       const float* __restrict__ Wc,      // [B,C,E]
                       const float* __restrict__ cw,      // [NG,C,C]
                       const float* __restrict__ cb,      // [C]
                       const float* __restrict__ Wd,      // [C,E]
                       const float* __restrict__ bd,      // [C]
                       float* __restrict__ out)           // [B,C]
{
    extern __shared__ char smem[];
    __bf16* Bhi  = (__bf16*)(smem + OFF_B0);          // [48 cols][320 K] transposed
    __bf16* Blo  = Bhi + 48 * 320;
    __bf16* cwb  = (__bf16*)(smem + OFF_B0);          // [55][20 c][32 k] (overlay, post-GEMM)
    __bf16* Gp   = (__bf16*)(smem + OFF_G);           // [1078][32]
    float*  Pm   = (float*)(smem + OFF_P);            // [1024][20]
    float*  invn = (float*)(smem + OFF_MISC);
    float*  redD = invn + 32;
    float*  redN = redD + 32;

    const int b    = blockIdx.x;
    const int tid  = threadIdx.x;
    const int lane = tid & 31;
    const int wid  = tid >> 5;        // 16 waves
    const int l16  = lane & 15;
    const int hi16 = lane >> 4;       // 0 = k 0-7,16-23 of each K tile, 1 = k 8-15,24-31

    // ---------- Phase 0a: zero padded-G, compute W_class row inv-norms ----------
    {
        uint* gz = (uint*)Gp;
        for (int i = tid; i < GPAD_ROWS * 32 / 2; i += 512) gz[i] = 0u;
    }
    for (int r = wid; r < CC; r += 16) {
        const float* wr = Wc + ((size_t)b * CC + r) * EE;
        float ss = 0.f;
        for (int k = lane; k < EE; k += 32) { float v = wr[k]; ss += v * v; }
#pragma unroll
        for (int off = 16; off > 0; off >>= 1) ss += __shfl_xor(ss, off, 32);
        if (lane == 0) invn[r] = rsqrtf(fmaxf(ss, 1e-12f));
    }
    __syncthreads();

    // ---------- Phase 0b: build B = [wn | W_dis | 0] as hi/lo bf16, col-major ----------
    for (int e = tid; e < 48 * 320; e += 512) {
        int col = e / 320, k = e - col * 320;
        float v = 0.f;
        if (k < EE) {
            if (col < CC)            v = Wc[((size_t)b * CC + col) * EE + k] * invn[col];
            else if (col < 2 * CC)   v = Wd[(size_t)(col - CC) * EE + k];
        }
        __bf16 h = (__bf16)v;
        Bhi[e] = h;
        Blo[e] = (__bf16)(v - (float)h);
    }
    __syncthreads();

    // ---------- Phase 1: R[1024x40] = x_emb[b] * B  (3-term bf16 WMMA), scatter G/P ----------
#pragma unroll 1
    for (int m = 0; m < 4; ++m) {
        const int Mb  = wid * 64 + m * 16;
        const int row = Mb + l16;
        const float* arow = x_emb + ((size_t)b * LL + row) * EE;
        const int off = hi16 ? 8 : 0;
        v8f acc[3]; acc[0] = (v8f)0.f; acc[1] = (v8f)0.f; acc[2] = (v8f)0.f;
        float ss = 0.f;

        // steady state: K tiles 0..8 are fully in-bounds (k<=287 < 300) -> no predication
#pragma unroll 1
        for (int kt = 0; kt < 9; ++kt) {
            const int k0 = kt * 32;
            const int ka = k0 + off, kb = k0 + 16 + off;
            __builtin_prefetch(arow + k0 + 32 + off, 0, 1);   // global_prefetch_b8, next K tile
            float4 f0 = *(const float4*)(arow + ka);
            float4 f1 = *(const float4*)(arow + ka + 4);
            float4 f2 = *(const float4*)(arow + kb);
            float4 f3 = *(const float4*)(arow + kb + 4);
            float xv[16] = { f0.x, f0.y, f0.z, f0.w, f1.x, f1.y, f1.z, f1.w,
                             f2.x, f2.y, f2.z, f2.w, f3.x, f3.y, f3.z, f3.w };
            v16bf ah, al;
#pragma unroll
            for (int i = 0; i < 16; ++i) {
                float v = xv[i]; ss += v * v;
                __bf16 h = (__bf16)v;
                ah[i] = h; al[i] = (__bf16)(v - (float)h);
            }
#pragma unroll
            for (int nt = 0; nt < 3; ++nt) {
                const int n = nt * 16 + l16;
                const __bf16* bhP = Bhi + n * 320 + k0 + (hi16 ? 8 : 0);
                const __bf16* blP = Blo + n * 320 + k0 + (hi16 ? 8 : 0);
                v16bf bh = join16(*(const v8bf*)bhP, *(const v8bf*)(bhP + 16));
                v16bf bl = join16(*(const v8bf*)blP, *(const v8bf*)(blP + 16));
                acc[nt] = __builtin_amdgcn_wmma_f32_16x16x32_bf16(false, ah, false, bh, (short)0, acc[nt], false, false);
                acc[nt] = __builtin_amdgcn_wmma_f32_16x16x32_bf16(false, ah, false, bl, (short)0, acc[nt], false, false);
                acc[nt] = __builtin_amdgcn_wmma_f32_16x16x32_bf16(false, al, false, bh, (short)0, acc[nt], false, false);
            }
        }

        // tail: K tile 9 (k0 = 288), elements k >= 300 are zero
        {
            const int k0 = 288;
            const int ka = k0 + off, kb = k0 + 16 + off;
            float4 f0 = ld4_pred(arow + ka,     ka     < EE);
            float4 f1 = ld4_pred(arow + ka + 4, ka + 4 < EE);
            float4 f2 = ld4_pred(arow + kb,     kb     < EE);
            float4 f3 = ld4_pred(arow + kb + 4, kb + 4 < EE);
            float xv[16] = { f0.x, f0.y, f0.z, f0.w, f1.x, f1.y, f1.z, f1.w,
                             f2.x, f2.y, f2.z, f2.w, f3.x, f3.y, f3.z, f3.w };
            v16bf ah, al;
#pragma unroll
            for (int i = 0; i < 16; ++i) {
                float v = xv[i]; ss += v * v;
                __bf16 h = (__bf16)v;
                ah[i] = h; al[i] = (__bf16)(v - (float)h);
            }
#pragma unroll
            for (int nt = 0; nt < 3; ++nt) {
                const int n = nt * 16 + l16;
                const __bf16* bhP = Bhi + n * 320 + k0 + (hi16 ? 8 : 0);
                const __bf16* blP = Blo + n * 320 + k0 + (hi16 ? 8 : 0);
                v16bf bh = join16(*(const v8bf*)bhP, *(const v8bf*)(bhP + 16));
                v16bf bl = join16(*(const v8bf*)blP, *(const v8bf*)(blP + 16));
                acc[nt] = __builtin_amdgcn_wmma_f32_16x16x32_bf16(false, ah, false, bh, (short)0, acc[nt], false, false);
                acc[nt] = __builtin_amdgcn_wmma_f32_16x16x32_bf16(false, ah, false, bl, (short)0, acc[nt], false, false);
                acc[nt] = __builtin_amdgcn_wmma_f32_16x16x32_bf16(false, al, false, bh, (short)0, acc[nt], false, false);
            }
        }

        // per-row scale = mask * rsqrt(max(mask^2 * sumsq, eps))
        ss += __shfl_xor(ss, 16, 32);
        const float mrow  = x_mask[(size_t)b * LL + row];
        const float scale = mrow * rsqrtf(fmaxf(mrow * mrow * ss, 1e-12f));
#pragma unroll
        for (int nt = 0; nt < 3; ++nt) {
            const int colc = nt * 16 + l16;
#pragma unroll
            for (int k = 0; k < 8; ++k) {
                const int rsrc = hi16 ? (k + 8) : k;
                const float sc = __shfl(scale, rsrc, 32);
                const int rrow = Mb + rsrc;
                const float v  = acc[nt][k];
                if (colc < CC)              Gp[(27 + rrow) * 32 + colc] = (__bf16)(v * sc);
                else if (colc < 2 * CC)     Pm[rrow * CC + (colc - CC)] = v;
            }
        }
    }
    __syncthreads();

    // ---------- Phase 1.5: conv weights -> bf16 LDS [w][c_out][k_in(pad32)] ----------
    for (int e = tid; e < NG * CC * 32; e += 512) {
        int w = e / (CC * 32), rem = e - w * (CC * 32);
        int c = rem >> 5, k = rem & 31;
        float v = (k < CC) ? cw[(size_t)w * CC * CC + k * CC + c] : 0.f;
        cwb[e] = (__bf16)v;
    }
    if (tid < 32) { redD[tid] = 0.f; redN[tid] = 0.f; }
    __syncthreads();

    // ---------- Phase 2: conv as 55 shifted 20x20 GEMMs (WMMA), fused softmax reduce ----------
    v8f a2[4][2];
#pragma unroll
    for (int m = 0; m < 4; ++m) { a2[m][0] = (v8f)0.f; a2[m][1] = (v8f)0.f; }

#pragma unroll 1
    for (int w55 = 0; w55 < NG; ++w55) {
        v16bf bf[2];
#pragma unroll
        for (int nt = 0; nt < 2; ++nt) {
            const int c = nt * 16 + l16;
            if (c < CC) {
                const __bf16* p = cwb + ((size_t)w55 * CC + c) * 32 + (hi16 ? 8 : 0);
                bf[nt] = join16(*(const v8bf*)p, *(const v8bf*)(p + 16));
            } else {
                bf[nt] = (v16bf)(__bf16)0.0f;
            }
        }
#pragma unroll
        for (int m = 0; m < 4; ++m) {
            const int srow = wid * 64 + m * 16 + l16 + w55;   // padded-G row
            const __bf16* gb = Gp + srow * 32;
            v8bf g0 = *(const v8bf*)(gb + (hi16 ? 8 : 0));
            v8bf g1 = *(const v8bf*)(gb + 16 + (hi16 ? 8 : 0));
            v16bf af = join16(g0, g1);
            a2[m][0] = __builtin_amdgcn_wmma_f32_16x16x32_bf16(false, af, false, bf[0], (short)0, a2[m][0], false, false);
            a2[m][1] = __builtin_amdgcn_wmma_f32_16x16x32_bf16(false, af, false, bf[1], (short)0, a2[m][1], false, false);
        }
    }

    // fused: relu -> exp -> mask -> per-column partial sums of ew and ew*P
    const int c0 = l16;                // 0..15, always valid
    const int c1 = 16 + l16;           // valid if < 20
    const float cb0 = cb[c0];
    const float cb1 = (c1 < CC) ? cb[c1] : 0.f;
    float den0 = 0.f, num0 = 0.f, den1 = 0.f, num1 = 0.f;
#pragma unroll
    for (int m = 0; m < 4; ++m) {
#pragma unroll
        for (int k = 0; k < 8; ++k) {
            const int rrow = wid * 64 + m * 16 + (hi16 ? (k + 8) : k);
            const float mk = x_mask[(size_t)b * LL + rrow];
            {
                float att = a2[m][0][k] + cb0;
                float e   = __expf(fmaxf(att, 0.f)) * mk;
                den0 += e;
                num0 += e * Pm[rrow * CC + c0];
            }
            if (c1 < CC) {
                float att = a2[m][1][k] + cb1;
                float e   = __expf(fmaxf(att, 0.f)) * mk;
                den1 += e;
                num1 += e * Pm[rrow * CC + c1];
            }
        }
    }
    atomicAdd(&redD[c0], den0);
    atomicAdd(&redN[c0], num0);
    if (c1 < CC) { atomicAdd(&redD[c1], den1); atomicAdd(&redN[c1], num1); }
    __syncthreads();

    // ---------- Phase 3: logits ----------
    if (tid < CC) out[(size_t)b * CC + tid] = redN[tid] / redD[tid] + bd[tid];
}

extern "C" void kernel_launch(void* const* d_in, const int* in_sizes, int n_in,
                              void* d_out, int out_size, void* d_ws, size_t ws_size,
                              hipStream_t stream) {
    (void)in_sizes; (void)n_in; (void)d_ws; (void)ws_size; (void)out_size;
    const float* x_emb  = (const float*)d_in[0];
    const float* x_mask = (const float*)d_in[1];
    const float* Wc     = (const float*)d_in[2];
    const float* cw     = (const float*)d_in[3];
    const float* cb     = (const float*)d_in[4];
    const float* Wd     = (const float*)d_in[5];
    const float* bd     = (const float*)d_in[6];
    float* out = (float*)d_out;

    leam_fused_kernel<<<BB, 512, SMEM_BYTES, stream>>>(
        x_emb, x_mask, Wc, cw, cb, Wd, bd, out);
}
